// GraphAttentionBlock_56985626083974
// MI455X (gfx1250) — compile-verified, matrix operand order checked by reading, not run on hardware
//
#include <hip/hip_runtime.h>
#include <stdint.h>

// ---------------------------------------------------------------------------
// Gated GraphConv (L=2, N=100000, D=128) for gfx1250 (MI455X).
//   - GEMMs: v_wmma_f32_16x16x32_bf16, fp32 accumulate.
//   - A-fragments built in-register from fp32 rows; fp32->bf16 pair packing
//     is a single v_perm_b32 (byte select) per 2 elements (truncation), or
//     v_cvt_pk_bf16_f32 where that builtin exists.
//   - Messages m stored bf16 -> edge gather traffic halved (410 MB/layer).
//   - Scatter-add: global_atomic_add_f32 (unsafeAtomicAdd), fp32 accumulate.
// Per-layer traffic ~1.5 GB -> ~130 us/layer floor @ 23.3 TB/s; GEMM FLOPs
// (46 GFLOP total) are negligible on the WMMA path. agg (51 MB) and m_bf
// (25.6 MB) are L2-resident (192 MB L2).
// ---------------------------------------------------------------------------

#define N_NODES 100000
#define N_EDGES 1600000
#define DIM     128
#define GATES   384   // 3*D
#define NLAYERS 2

typedef __attribute__((ext_vector_type(16))) __bf16 v16bf;
typedef __attribute__((ext_vector_type(2)))  __bf16 v2bf;
typedef __attribute__((ext_vector_type(8)))  float  v8f;

union Frag16 { v16bf v; uint4 q[2]; };

__device__ __forceinline__ unsigned int f2bf_bits(float f) {
  unsigned int u = __float_as_uint(f);
  return (u + 0x7FFFu + ((u >> 16) & 1u)) >> 16;   // round-to-nearest-even
}
__device__ __forceinline__ float bf2f(unsigned int lo16) {
  return __uint_as_float(lo16 << 16);
}

// RNE pack (used for one-time weight conversion).
__device__ __forceinline__ unsigned int pack2_rne(float lo, float hi) {
  return f2bf_bits(lo) | (f2bf_bits(hi) << 16);
}

// Fast pack for per-iteration fragment building: one VALU op per pair.
// Prefers the HW packed convert; else v_perm_b32 byte-select {hi.b3,hi.b2,
// lo.b3,lo.b2} (truncation, <=1 ulp-of-bf16 vs RNE).
__device__ __forceinline__ unsigned int pack2_fast(float lo, float hi) {
#if __has_builtin(__builtin_amdgcn_cvt_pk_bf16_f32)
  union { v2bf v; unsigned int u; } c;
  c.v = __builtin_amdgcn_cvt_pk_bf16_f32(lo, hi);
  return c.u;
#else
  return __builtin_amdgcn_perm(__float_as_uint(hi), __float_as_uint(lo),
                               0x07060302u);
#endif
}

__device__ __forceinline__ float sigmoidf_(float x) {
  return 1.0f / (1.0f + __expf(-x));
}

// 16x32 bf16 fragment per ISA layout, from a pre-converted bf16 row (B side):
//  lane = half*16 + l16; element j: K = (j<8?0:16) + half*8 + (j&7)
__device__ __forceinline__ v16bf load_frag_bf16(const unsigned short* __restrict__ row,
                                                int kbase, int half) {
  Frag16 f;
  f.q[0] = *reinterpret_cast<const uint4*>(row + kbase + half * 8);
  f.q[1] = *reinterpret_cast<const uint4*>(row + kbase + 16 + half * 8);
  return f.v;
}

// Same fragment built from an fp32 row with in-register bf16 packing (A side).
__device__ __forceinline__ v16bf load_frag_f32(const float* __restrict__ row,
                                               int kbase, int half) {
  const float4* p0 = reinterpret_cast<const float4*>(row + kbase + half * 8);
  const float4* p1 = reinterpret_cast<const float4*>(row + kbase + 16 + half * 8);
  float4 a0 = p0[0], a1 = p0[1];
  float4 b0 = p1[0], b1 = p1[1];
  Frag16 f;
  f.q[0] = make_uint4(pack2_fast(a0.x, a0.y), pack2_fast(a0.z, a0.w),
                      pack2_fast(a1.x, a1.y), pack2_fast(a1.z, a1.w));
  f.q[1] = make_uint4(pack2_fast(b0.x, b0.y), pack2_fast(b0.z, b0.w),
                      pack2_fast(b1.x, b1.y), pack2_fast(b1.z, b1.w));
  return f.v;
}

// ---------------------------------------------------------------------------
// Small prep kernels
// ---------------------------------------------------------------------------
__global__ __launch_bounds__(256) void f32_to_bf16(const float* __restrict__ src,
                                                   unsigned short* __restrict__ dst,
                                                   int n4) {
  int i = blockIdx.x * 256 + threadIdx.x;
  if (i >= n4) return;
  float4 v = reinterpret_cast<const float4*>(src)[i];
  reinterpret_cast<uint2*>(dst)[i] = make_uint2(pack2_rne(v.x, v.y), pack2_rne(v.z, v.w));
}

// weight[i] is [K=128][N=128] row-major; store dst[n*128+k] = bf16(src[k*128+n])
__global__ __launch_bounds__(256) void transpose_to_bf16(const float* __restrict__ src,
                                                         unsigned short* __restrict__ dst) {
  int i = blockIdx.x * 256 + threadIdx.x;  // 16384 total
  int k = i >> 7, n = i & 127;
  dst[n * DIM + k] = (unsigned short)f2bf_bits(src[k * DIM + n]);
}

__global__ __launch_bounds__(256) void zero_f32(float* __restrict__ p, int n4) {
  int i = blockIdx.x * 256 + threadIdx.x;
  if (i < n4) reinterpret_cast<float4*>(p)[i] = make_float4(0.f, 0.f, 0.f, 0.f);
}

// ---------------------------------------------------------------------------
// m = h @ W  (m stored bf16): 16 rows x 128 cols per block; 8 waves, one
// 16x16 WMMA tile each; A converted fp32->bf16 in-register.
// ---------------------------------------------------------------------------
__global__ __launch_bounds__(256) void gemm_hW(const float* __restrict__ h,
                                               const unsigned short* __restrict__ wtbf,
                                               unsigned short* __restrict__ mbf) {
  const int wave = threadIdx.x >> 5;
  const int lane = threadIdx.x & 31;
  const int half = lane >> 4;
  const int l16  = lane & 15;
  const int row0 = blockIdx.x << 4;
  const int col0 = wave << 4;

  const float*          arow = h    + (size_t)(row0 + l16) * DIM;
  const unsigned short* brow = wtbf + (size_t)(col0 + l16) * DIM;

  v8f acc = {};
#pragma unroll
  for (int kb = 0; kb < DIM; kb += 32) {
    v16bf a = load_frag_f32(arow, kb, half);
    v16bf b = load_frag_bf16(brow, kb, half);
    acc = __builtin_amdgcn_wmma_f32_16x16x32_bf16(false, a, false, b,
                                                  (short)0, acc, false, false);
  }
#pragma unroll
  for (int r = 0; r < 8; ++r) {
    int m = r + half * 8;   // D layout: VGPR r -> row r (lanes 0-15) / r+8 (lanes 16-31)
    mbf[(size_t)(row0 + m) * DIM + col0 + l16] = (unsigned short)f2bf_bits(acc[r]);
  }
}

// ---------------------------------------------------------------------------
// Edge gather (bf16, 256 B/edge) + scale + fp32 scatter-add. One wave/edge.
// ---------------------------------------------------------------------------
__global__ __launch_bounds__(256) void edge_scatter(const int* __restrict__ ei,
                                                    const float* __restrict__ ew,
                                                    const unsigned short* __restrict__ mbf,
                                                    float* __restrict__ agg) {
  const int wave = threadIdx.x >> 5;
  const int lane = threadIdx.x & 31;
  const size_t e = (size_t)blockIdx.x * 8 + wave;
  const int s = ei[e];
  const int d = ei[(size_t)N_EDGES + e];
  const float w = ew[e];
  const uint2 v = reinterpret_cast<const uint2*>(mbf + (size_t)s * DIM)[lane];
  float* dst = agg + (size_t)d * DIM + lane * 4;
  unsafeAtomicAdd(dst + 0, bf2f(v.x & 0xFFFFu) * w);
  unsafeAtomicAdd(dst + 1, bf2f(v.x >> 16) * w);
  unsafeAtomicAdd(dst + 2, bf2f(v.y & 0xFFFFu) * w);
  unsafeAtomicAdd(dst + 3, bf2f(v.y >> 16) * w);
}

// ---------------------------------------------------------------------------
// Fused GRU cell: gi = agg @ w_ih^T + b_ih, gh = h @ w_hh^T + b_hh,
// gates via 48 KB LDS, then elementwise update. 16 nodes per block, 8 waves,
// each wave owns 3 gi-tiles + 3 gh-tiles (192 WMMAs/block). A-fragments are
// packed fp32->bf16 in-register from agg / hprev.
// ---------------------------------------------------------------------------
__global__ __launch_bounds__(256) void gru_fused(const float* __restrict__ aggf,
                                                 const float* __restrict__ hprev,
                                                 const unsigned short* __restrict__ wihbf,
                                                 const unsigned short* __restrict__ whhbf,
                                                 const float* __restrict__ bih,
                                                 const float* __restrict__ bhh,
                                                 float* __restrict__ hout) {
  __shared__ float gi_s[16 * GATES];   // 24 KB
  __shared__ float gh_s[16 * GATES];   // 24 KB

  const int wave = threadIdx.x >> 5;
  const int lane = threadIdx.x & 31;
  const int half = lane >> 4;
  const int l16  = lane & 15;
  const int row0 = blockIdx.x << 4;

  const float* arow = aggf  + (size_t)(row0 + l16) * DIM;
  const float* hrow = hprev + (size_t)(row0 + l16) * DIM;

  v8f gi_acc[3], gh_acc[3];
#pragma unroll
  for (int j = 0; j < 3; ++j) { v8f z = {}; gi_acc[j] = z; gh_acc[j] = z; }

#pragma unroll
  for (int kb = 0; kb < DIM; kb += 32) {
    v16bf aa = load_frag_f32(arow, kb, half);
    v16bf ah = load_frag_f32(hrow, kb, half);
#pragma unroll
    for (int j = 0; j < 3; ++j) {
      const int col0 = (wave * 3 + j) << 4;
      v16bf bi = load_frag_bf16(wihbf + (size_t)(col0 + l16) * DIM, kb, half);
      gi_acc[j] = __builtin_amdgcn_wmma_f32_16x16x32_bf16(false, aa, false, bi,
                                                          (short)0, gi_acc[j], false, false);
      v16bf bh = load_frag_bf16(whhbf + (size_t)(col0 + l16) * DIM, kb, half);
      gh_acc[j] = __builtin_amdgcn_wmma_f32_16x16x32_bf16(false, ah, false, bh,
                                                          (short)0, gh_acc[j], false, false);
    }
  }

#pragma unroll
  for (int j = 0; j < 3; ++j) {
    const int col0 = (wave * 3 + j) << 4;
#pragma unroll
    for (int r = 0; r < 8; ++r) {
      int m = r + half * 8;
      gi_s[m * GATES + col0 + l16] = gi_acc[j][r];
      gh_s[m * GATES + col0 + l16] = gh_acc[j][r];
    }
  }
  __syncthreads();

  // 16 rows x 128 features = 2048 elems; 256 threads x 8
#pragma unroll
  for (int it = 0; it < 8; ++it) {
    int idx = threadIdx.x + it * 256;
    int rr = idx >> 7;
    int d  = idx & 127;
    float ir  = gi_s[rr * GATES + d]       + bih[d];
    float iz  = gi_s[rr * GATES + 128 + d] + bih[128 + d];
    float in_ = gi_s[rr * GATES + 256 + d] + bih[256 + d];
    float hr  = gh_s[rr * GATES + d]       + bhh[d];
    float hz  = gh_s[rr * GATES + 128 + d] + bhh[128 + d];
    float hn  = gh_s[rr * GATES + 256 + d] + bhh[256 + d];
    float r = sigmoidf_(ir + hr);
    float z = sigmoidf_(iz + hz);
    float n = tanhf(in_ + r * hn);
    float hp = hprev[(size_t)(row0 + rr) * DIM + d];
    hout[(size_t)(row0 + rr) * DIM + d] = (1.0f - z) * n + z * hp;
  }
}

// ---------------------------------------------------------------------------
// Host-side launch sequence (graph-capture safe: kernels only, on `stream`).
// ---------------------------------------------------------------------------
extern "C" void kernel_launch(void* const* d_in, const int* in_sizes, int n_in,
                              void* d_out, int out_size, void* d_ws, size_t ws_size,
                              hipStream_t stream) {
  (void)in_sizes; (void)n_in; (void)out_size; (void)ws_size;

  const float* x   = (const float*)d_in[0];
  const int*   ei  = (const int*)d_in[1];   // edge_index flat [2, E]
  const float* ew  = (const float*)d_in[2];
  const float* wgt = (const float*)d_in[3]; // [L, 128, 128]
  const float* wih = (const float*)d_in[4]; // [384, 128]
  const float* whh = (const float*)d_in[5]; // [384, 128]
  const float* bih = (const float*)d_in[6]; // [384]
  const float* bhh = (const float*)d_in[7]; // [384]

  char* ws = (char*)d_ws;
  size_t off = 0;
  unsigned short* wih_bf = (unsigned short*)(ws + off); off += (size_t)GATES * DIM * 2;
  unsigned short* whh_bf = (unsigned short*)(ws + off); off += (size_t)GATES * DIM * 2;
  unsigned short* wt_bf  = (unsigned short*)(ws + off); off += (size_t)DIM * DIM * 2;
  off = (off + 255) & ~(size_t)255;
  unsigned short* m_bf   = (unsigned short*)(ws + off); off += (size_t)N_NODES * DIM * 2;
  float* agg_f = (float*)(ws + off); off += (size_t)N_NODES * DIM * 4;
  float* h1    = (float*)(ws + off); off += (size_t)N_NODES * DIM * 4;

  const int ND4 = N_NODES * DIM / 4;   // 3,200,000 float4s
  const int GW4 = GATES * DIM / 4;     // 12,288

  // GRU weights to bf16 once (already [out,in] == B^T layout for m @ W^T)
  f32_to_bf16<<<(GW4 + 255) / 256, 256, 0, stream>>>(wih, wih_bf, GW4);
  f32_to_bf16<<<(GW4 + 255) / 256, 256, 0, stream>>>(whh, whh_bf, GW4);

  for (int layer = 0; layer < NLAYERS; ++layer) {
    const float* hcur = (layer == 0) ? x : h1;
    float* hnext = (layer == NLAYERS - 1) ? (float*)d_out : h1;

    // W[layer] -> transposed bf16 (column-major so B loads are contiguous)
    transpose_to_bf16<<<DIM * DIM / 256, 256, 0, stream>>>(wgt + (size_t)layer * DIM * DIM, wt_bf);
    // m = h @ W   (bf16 output; A packed in-register)
    gemm_hW<<<N_NODES / 16, 256, 0, stream>>>(hcur, wt_bf, m_bf);
    // agg = segment_sum(m[src] * ew, dst)
    zero_f32<<<(ND4 + 255) / 256, 256, 0, stream>>>(agg_f, ND4);
    edge_scatter<<<N_EDGES / 8, 256, 0, stream>>>(ei, ew, m_bf, agg_f);
    // fused GRU cell (reads agg/h fp32, packs fragments in-register)
    gru_fused<<<N_NODES / 16, 256, 0, stream>>>(agg_f, hcur, wih_bf, whh_bf,
                                                bih, bhh, hnext);
  }
}
